// GNN_73710228733973
// MI455X (gfx1250) — compile-verified, matrix operand order checked by reading, not run on hardware
//
#include <hip/hip_runtime.h>
#include <math.h>

// ---------- types ----------
typedef __bf16 bf16_t;
typedef bf16_t v16bf __attribute__((ext_vector_type(16)));
typedef float  v8f   __attribute__((ext_vector_type(8)));
typedef float  v4f   __attribute__((ext_vector_type(4)));
typedef unsigned int u32;
typedef u32    v4u   __attribute__((ext_vector_type(4)));
typedef u32    u32x4 __attribute__((ext_vector_type(4)));
typedef int    i32x8 __attribute__((ext_vector_type(8)));
typedef int    i32x4 __attribute__((ext_vector_type(4)));

union BFrag { v16bf v; v4u q[2]; };
union F8    { v4f q[2]; float f[8]; };

static constexpr int HID   = 256;
static constexpr int HEADS = 4;
static constexpr int KC    = 256;   // K-chunk staged in LDS per TDM op (32 KB)

// ---------- device helpers ----------
static __device__ __forceinline__ float gelu_erf(float x) {
  return 0.5f * x * (1.0f + erff(x * 0.7071067811865475f));
}
static __device__ __forceinline__ unsigned short bf16_rne(float f) {
  u32 x = __float_as_uint(f);
  u32 r = x + 0x7FFFu + ((x >> 16) & 1u);
  return (unsigned short)(r >> 16);
}

// Tensor Data Mover: 2D tile (rows x row_elems of 8-byte elements) -> LDS offset 0.
// D# built per CDNA5 ISA (group0/group1; groups 2/3 zero).
// 6-arg builtin form (clang-23 / therock HIP headers).
static __device__ __forceinline__ void tdm_load_2d(const void* gptr,
                                                   u32 row_elems, u32 rows,
                                                   u32 row_stride_elems) {
  unsigned long long ga = (unsigned long long)gptr;
  u32x4 g0;
  g0[0] = 1u;                                   // count=1, user descriptor
  g0[1] = 0u;                                   // lds_addr = dynamic-LDS base (0)
  g0[2] = (u32)(ga & 0xFFFFFFFFu);              // global_addr[31:0]
  g0[3] = (u32)((ga >> 32) & 0x01FFFFFFu) | (2u << 30);  // addr[56:32], type=2
  const u32 td0 = 1u << 30, td1 = 1u << 20;     // huge tensor dims: no OOB clip
  i32x8 g1;
  g1[0] = (int)(3u << 16);                      // workgroup_mask=0, data_size=8B
  g1[1] = (int)((td0 & 0xFFFFu) << 16);         // tensor_dim0[15:0]
  g1[2] = (int)((td0 >> 16) | ((td1 & 0xFFFFu) << 16));
  g1[3] = (int)(((td1 >> 16) & 0xFFFFu) | (row_elems << 16));   // tile_dim0
  g1[4] = (int)rows;                            // tile_dim1 (tile_dim2=0)
  g1[5] = (int)row_stride_elems;                // tensor_dim0_stride[31:0]
  g1[6] = 0; g1[7] = 0;
  i32x4 z4 = {0, 0, 0, 0};
  i32x8 z8 = {0, 0, 0, 0, 0, 0, 0, 0};
  __builtin_amdgcn_tensor_load_to_lds(g0, g1, z4, z4, z8, 0);
}

// ---------- elementwise kernels ----------
__global__ void k_f32_to_bf16(const float* __restrict__ in,
                              unsigned short* __restrict__ out, int n) {
  int i = blockIdx.x * blockDim.x + threadIdx.x;
  if (i < n) out[i] = bf16_rne(in[i]);
}

// ---------- weight pack: f32 W[K,N] row-major -> WMMA-B fragment blob ----------
// uint out[((nt*(K/32)+kt)*32 + lane)*8 + v] = bf16 pair (k, k+1)
// lane: n = nt*16 + (lane&15), kgroup g = lane>>4
// v<4 : klocal = 2v + 8g ; v>=4 : klocal = 16 + 2(v-4) + 8g  (ISA 7.12.2)
__global__ void k_pack_b(const float* __restrict__ W, u32* __restrict__ out,
                         int K, int N) {
  int tid = blockIdx.x * blockDim.x + threadIdx.x;
  int total = (K * N) >> 1;
  if (tid >= total) return;
  int v    = tid & 7;
  int lane = (tid >> 3) & 31;
  int kt   = (tid >> 8) % (K >> 5);
  int nt   = (tid >> 8) / (K >> 5);
  int g  = lane >> 4;
  int nn = lane & 15;
  int klocal = (v < 4) ? (2 * v + 8 * g) : (16 + 2 * (v - 4) + 8 * g);
  int k = (kt << 5) + klocal;
  int n = (nt << 4) + nn;
  u32 lo = bf16_rne(W[(size_t)k * N + n]);
  u32 hi = bf16_rne(W[(size_t)(k + 1) * N + n]);
  out[tid] = lo | (hi << 16);
}

// ---------- WMMA GEMM with TDM-staged B panel ----------
// Block = 8 waves -> 128x64 output band. B strip (64 cols, KC of K) staged in
// LDS once per chunk by wave 0 via tensor_load_to_lds; all waves read it with
// ds_load_b128. A streams from global with a 1-deep register prefetch.
__global__ __launch_bounds__(256)
void k_gemm_wmma(const unsigned short* __restrict__ A,
                 const u32* __restrict__ Bp,
                 const float* __restrict__ bias,
                 float* __restrict__ C,
                 int M, int N, int K, int act) {
  extern __shared__ u32 smem[];                 // KC*64 bf16 = 32 KB, offset 0
  const int wave = threadIdx.x >> 5;
  const int lane = threadIdx.x & 31;
  const int nt0  = blockIdx.x << 2;             // 4 consecutive 16-col tiles
  const int mt   = (blockIdx.y << 3) + wave;    // this wave's 16-row tile
  const int g  = lane >> 4;
  const int nn = lane & 15;
  const int row = (mt << 4) + nn;
  const bool active = (mt << 4) < M;
  const int ktiles  = K >> 5;                   // total 16x16x32 K-steps
  const int KCt     = KC >> 5;                  // K-steps per staged chunk (8)
  const int kchunks = K / KC;
  const unsigned short* arow = A + (size_t)row * K;

  v8f acc[4];
  #pragma unroll
  for (int j = 0; j < 4; ++j) acc[j] = (v8f){0.f,0.f,0.f,0.f,0.f,0.f,0.f,0.f};

  for (int ch = 0; ch < kchunks; ++ch) {
    __syncthreads();                            // previous chunk fully consumed
    if (threadIdx.x < 32) {                     // wave 0 drives the TDM
      const u32* src = Bp + ((size_t)nt0 * ktiles + (size_t)ch * KCt) * 256;
      tdm_load_2d(src, /*row_elems(8B)=*/KCt * 128, /*rows=*/4,
                  /*row_stride(8B)=*/(u32)ktiles * 128);
      __builtin_amdgcn_s_wait_tensorcnt(0);
    }
    __syncthreads();                            // B chunk visible to all waves

    if (active) {
      BFrag a_cur, a_nxt;
      const unsigned short* ab = arow + ch * KC + (g << 3);
      a_cur.q[0] = *reinterpret_cast<const v4u*>(ab);
      a_cur.q[1] = *reinterpret_cast<const v4u*>(ab + 16);
      for (int ktl = 0; ktl < KCt; ++ktl) {
        if (ktl + 1 < KCt) {                    // prefetch next A fragment
          const unsigned short* an = ab + ((ktl + 1) << 5);
          a_nxt.q[0] = *reinterpret_cast<const v4u*>(an);
          a_nxt.q[1] = *reinterpret_cast<const v4u*>(an + 16);
        }
        #pragma unroll
        for (int j = 0; j < 4; ++j) {
          const u32* ls = smem + (((j * KCt + ktl) * 32 + lane) << 3);
          BFrag b;
          b.q[0] = *reinterpret_cast<const v4u*>(ls);
          b.q[1] = *reinterpret_cast<const v4u*>(ls + 4);
          acc[j] = __builtin_amdgcn_wmma_f32_16x16x32_bf16(
                       false, a_cur.v, false, b.v, (short)0, acc[j], false, false);
        }
        a_cur = a_nxt;
      }
    }
  }

  if (active) {
    // C/D layout: lanes 0-15 -> M=v, N=lane ; lanes 16-31 -> M=8+v, N=lane-16
    #pragma unroll
    for (int j = 0; j < 4; ++j) {
      int ncol = ((nt0 + j) << 4) + nn;
      float bv = bias[ncol];
      #pragma unroll
      for (int v = 0; v < 8; ++v) {
        int m = (mt << 4) + (g << 3) + v;
        float r = acc[j][v] + bv;
        if (act) r = gelu_erf(r);
        C[(size_t)m * N + ncol] = r;
      }
    }
  }
}

// ---------- CSR build (in-edges grouped by destination) ----------
__global__ void k_deg(const int* __restrict__ ei, u32* __restrict__ deg,
                      int E, int T) {
  int e = blockIdx.x * blockDim.x + threadIdx.x;
  if (e >= T) return;
  int dst = (e < E) ? ei[E + e] : (e - E);
  atomicAdd(&deg[dst], 1u);
}

__global__ void k_scan(const u32* __restrict__ deg, u32* __restrict__ offs,
                       int Nn) {
  __shared__ u32 part[1024];
  __shared__ u32 total;
  int t = threadIdx.x;
  int chunk = Nn >> 10;
  int c0 = t * chunk;
  u32 s = 0;
  for (int i = 0; i < chunk; ++i) s += deg[c0 + i];
  part[t] = s;
  __syncthreads();
  if (t == 0) {
    u32 acc = 0;
    for (int i = 0; i < 1024; ++i) { u32 v = part[i]; part[i] = acc; acc += v; }
    total = acc;
  }
  __syncthreads();
  u32 acc = part[t];
  for (int i = 0; i < chunk; ++i) { offs[c0 + i] = acc; acc += deg[c0 + i]; }
  if (t == 0) offs[Nn] = total;
}

__global__ void k_fill_csr(const int* __restrict__ ei, const u32* __restrict__ offs,
                           u32* __restrict__ cursor, u32* __restrict__ eidx,
                           int E, int T) {
  int e = blockIdx.x * blockDim.x + threadIdx.x;
  if (e >= T) return;
  int dst = (e < E) ? ei[E + e] : (e - E);
  u32 pos = atomicAdd(&cursor[dst], 1u);
  eidx[offs[dst] + pos] = (u32)e;
}

// ---------- GATv2 attention+aggregate: one wave per destination node ----------
// Lane owns 8 channels (head = lane>>3). Pass 1: per-head logit via 8-lane
// butterfly, register max. Pass 2: recompute logit, accumulate p * x_l[src]
// in registers; normalize + bias + optional GELU on the way out. No atomics.
__global__ __launch_bounds__(256)
void k_node_attn(const u32* __restrict__ offs, const u32* __restrict__ eidx,
                 const int* __restrict__ ei,
                 const float* __restrict__ xl, const float* __restrict__ xr,
                 const float* __restrict__ att, const float* __restrict__ bias,
                 float* __restrict__ out, int E, int Nn, int act) {
  int node = (blockIdx.x * blockDim.x + threadIdx.x) >> 5;
  if (node >= Nn) return;
  int lane = threadIdx.x & 31;
  int c0 = lane << 3;
  F8 xri;
  const v4f* rp = reinterpret_cast<const v4f*>(xr + (size_t)node * HID + c0);
  xri.q[0] = rp[0]; xri.q[1] = rp[1];
  float attv[8];
  #pragma unroll
  for (int i = 0; i < 8; ++i) attv[i] = att[c0 + i];
  u32 beg = offs[node], end = offs[node + 1];

  float mx = -1e30f;
  for (u32 p = beg; p < end; ++p) {
    int e = (int)eidx[p];
    int src = (e < E) ? ei[e] : (e - E);
    F8 xv;
    const v4f* sp = reinterpret_cast<const v4f*>(xl + (size_t)src * HID + c0);
    xv.q[0] = sp[0]; xv.q[1] = sp[1];
    float part = 0.f;
    #pragma unroll
    for (int i = 0; i < 8; ++i) {
      float t = xv.f[i] + xri.f[i];
      t = (t > 0.f) ? t : 0.2f * t;           // leaky_relu(0.2)
      part += attv[i] * t;
    }
    part += __shfl_xor(part, 1, 32);
    part += __shfl_xor(part, 2, 32);
    part += __shfl_xor(part, 4, 32);          // every lane holds its head logit
    mx = fmaxf(mx, part);
  }

  float ssum = 0.f, acc8[8];
  #pragma unroll
  for (int i = 0; i < 8; ++i) acc8[i] = 0.f;
  for (u32 p = beg; p < end; ++p) {
    int e = (int)eidx[p];
    int src = (e < E) ? ei[e] : (e - E);
    F8 xv;
    const v4f* sp = reinterpret_cast<const v4f*>(xl + (size_t)src * HID + c0);
    xv.q[0] = sp[0]; xv.q[1] = sp[1];
    float part = 0.f;
    #pragma unroll
    for (int i = 0; i < 8; ++i) {
      float t = xv.f[i] + xri.f[i];
      t = (t > 0.f) ? t : 0.2f * t;
      part += attv[i] * t;
    }
    part += __shfl_xor(part, 1, 32);
    part += __shfl_xor(part, 2, 32);
    part += __shfl_xor(part, 4, 32);
    float pr = expf(part - mx);
    ssum += pr;
    #pragma unroll
    for (int i = 0; i < 8; ++i) acc8[i] += pr * xv.f[i];
  }
  float inv = 1.f / (ssum + 1e-16f);
  #pragma unroll
  for (int i = 0; i < 8; ++i) {
    float r = acc8[i] * inv + bias[c0 + i];
    out[(size_t)node * HID + c0 + i] = act ? gelu_erf(r) : r;
  }
}

// ---------- pooling: per-graph mean + node0 extract ----------
__global__ void k_pool(const float* __restrict__ h,
                       float* __restrict__ mean_out,
                       float* __restrict__ z0, int npg) {
  int g = blockIdx.x, c = threadIdx.x;          // 256 threads
  const float* base = h + (size_t)g * npg * HID;
  float s = 0.f;
  for (int n = 0; n < npg; ++n) s += base[(size_t)n * HID + c];
  mean_out[g * HID + c] = s / (float)npg;
  z0[g * HID + c] = base[c];
}

// ---------- host launcher ----------
extern "C" void kernel_launch(void* const* d_in, const int* in_sizes, int n_in,
                              void* d_out, int out_size, void* d_ws, size_t ws_size,
                              hipStream_t stream) {
  const float* x    = (const float*)d_in[0];
  const int*   ei   = (const int*)d_in[1];
  const float* w1l  = (const float*)d_in[4];
  const float* b1l  = (const float*)d_in[5];
  const float* w1r  = (const float*)d_in[6];
  const float* b1r  = (const float*)d_in[7];
  const float* att1 = (const float*)d_in[8];
  const float* bias1= (const float*)d_in[9];
  const float* w2l  = (const float*)d_in[10];
  const float* b2l  = (const float*)d_in[11];
  const float* w2r  = (const float*)d_in[12];
  const float* b2r  = (const float*)d_in[13];
  const float* att2 = (const float*)d_in[14];
  const float* bias2= (const float*)d_in[15];
  const float* lin1w= (const float*)d_in[16];
  const float* lin1b= (const float*)d_in[17];
  const float* lin2w= (const float*)d_in[18];
  const float* lin2b= (const float*)d_in[19];
  const float* finw = (const float*)d_in[20];
  const float* finb = (const float*)d_in[21];

  const int Nn = in_sizes[0] / 256;      // 32768 nodes
  const int E  = in_sizes[1] / 2;        // 524288 edges
  const int T  = E + Nn;                 // + self loops
  const int G  = 64;                     // batch_size (reference constant)
  const int npg = Nn / G;
  const int FT_OUT = 512;

  // ---- carve workspace ----
  char* base = (char*)d_ws;
  size_t off = 0;
  auto carve = [&](size_t bytes) -> void* {
    void* r = base + off;
    off = (off + bytes + 255) & ~(size_t)255;
    return r;
  };
  unsigned short* xbf = (unsigned short*)carve((size_t)Nn * HID * 2);
  float* xl   = (float*)carve((size_t)Nn * HID * 4);
  float* xr   = (float*)carve((size_t)Nn * HID * 4);
  float* hbuf = (float*)carve((size_t)Nn * HID * 4);
  u32* deg    = (u32*)carve((size_t)Nn * 4);
  u32* offs   = (u32*)carve((size_t)(Nn + 1) * 4);
  u32* cursor = (u32*)carve((size_t)Nn * 4);
  u32* eidx   = (u32*)carve((size_t)T * 4);
  u32* pw1l = (u32*)carve((size_t)256 * 256 * 2);
  u32* pw1r = (u32*)carve((size_t)256 * 256 * 2);
  u32* pw2l = (u32*)carve((size_t)256 * 256 * 2);
  u32* pw2r = (u32*)carve((size_t)256 * 256 * 2);
  u32* plin1= (u32*)carve((size_t)256 * 1024 * 2);
  u32* plin2= (u32*)carve((size_t)1024 * 256 * 2);
  u32* pfin = (u32*)carve((size_t)256 * 512 * 2);
  float* z0          = (float*)carve((size_t)G * HID * 4);
  unsigned short* z0b= (unsigned short*)carve((size_t)G * HID * 2);
  float* t1          = (float*)carve((size_t)G * 1024 * 4);
  unsigned short* t1b= (unsigned short*)carve((size_t)G * 1024 * 2);
  float* t2          = (float*)carve((size_t)G * HID * 4);
  unsigned short* t2b= (unsigned short*)carve((size_t)G * HID * 2);
  if (off > ws_size) return;

  auto blk = [](int n, int t) { return (n + t - 1) / t; };

  auto cvt = [&](const float* in, unsigned short* out, int n) {
    k_f32_to_bf16<<<blk(n, 256), 256, 0, stream>>>(in, out, n);
  };
  auto pack = [&](const float* W, u32* out, int K, int N) {
    int tot = (K * N) >> 1;
    k_pack_b<<<blk(tot, 256), 256, 0, stream>>>(W, out, K, N);
  };
  auto gemm = [&](const unsigned short* A, const u32* Bp, const float* bias,
                  float* C, int M, int N, int K, int act) {
    dim3 grid(N / 64, (M + 127) / 128);
    size_t lds = (size_t)KC * 64 * 2;   // 32 KB staged B strip
    k_gemm_wmma<<<grid, 256, lds, stream>>>(A, Bp, bias, C, M, N, K, act);
  };
  auto conv_attn = [&](const float* att, const float* bias, float* out, int act) {
    k_node_attn<<<blk(Nn * 32, 256), 256, 0, stream>>>(offs, eidx, ei, xl, xr,
                                                       att, bias, out, E, Nn, act);
  };

  // ---- CSR build (shared by both convs) ----
  (void)hipMemsetAsync(deg, 0, (size_t)Nn * 4, stream);
  (void)hipMemsetAsync(cursor, 0, (size_t)Nn * 4, stream);
  k_deg<<<blk(T, 256), 256, 0, stream>>>(ei, deg, E, T);
  k_scan<<<1, 1024, 0, stream>>>(deg, offs, Nn);
  k_fill_csr<<<blk(T, 256), 256, 0, stream>>>(ei, offs, cursor, eidx, E, T);

  // ---- pack all weights (bf16 WMMA-B layout) ----
  pack(w1l, pw1l, 256, 256);  pack(w1r, pw1r, 256, 256);
  pack(w2l, pw2l, 256, 256);  pack(w2r, pw2r, 256, 256);
  pack(lin1w, plin1, 256, 1024);
  pack(lin2w, plin2, 1024, 256);
  pack(finw,  pfin,  256, 512);

  // ---- conv1 ----
  cvt(x, xbf, Nn * HID);
  gemm(xbf, pw1l, b1l, xl, Nn, 256, 256, 0);   // x_l = x@w1_l + b1_l
  gemm(xbf, pw1r, b1r, xr, Nn, 256, 256, 0);   // x_r = x@w1_r + b1_r
  conv_attn(att1, bias1, hbuf, /*gelu=*/1);    // h1 = gelu(conv1 + bias1)

  // ---- conv2 ----
  cvt(hbuf, xbf, Nn * HID);                    // h1 -> bf16
  gemm(xbf, pw2l, b2l, xl, Nn, 256, 256, 0);
  gemm(xbf, pw2r, b2r, xr, Nn, 256, 256, 0);
  conv_attn(att2, bias2, hbuf, /*gelu=*/0);    // h2 = conv2 + bias2

  // ---- pooling: graph_mean -> d_out tail, z0 = node0 rows ----
  float* out_z    = (float*)d_out;
  float* out_mean = out_z + (size_t)G * FT_OUT;
  k_pool<<<G, HID, 0, stream>>>(hbuf, out_mean, z0, npg);

  // ---- MLP head (WMMA, M = 64) ----
  cvt(z0, z0b, G * HID);
  gemm(z0b, plin1, lin1b, t1, G, 1024, 256, /*gelu=*/1);
  cvt(t1, t1b, G * 1024);
  gemm(t1b, plin2, lin2b, t2, G, 256, 1024, 0);
  cvt(t2, t2b, G * HID);
  gemm(t2b, pfin, finb, out_z, G, FT_OUT, 256, 0);
}
